// GCNAggregator_386547056898
// MI455X (gfx1250) — matrix operand order repaired and checked by reading, not compiled, not run on hardware
//
#include <hip/hip_runtime.h>

// ---------------------------------------------------------------------------
// GCN aggregator for MI455X (gfx1250, wave32, WMMA).
//
// symnorm(M) @ E  ==  diag(rsqrt(rowsum)) * ( M @ (diag(rsqrt(colsum)) * E) )
// M is a 0/1 mask -> exact in f16. E_scaled is split into f16 hi+lo so that
// two v_wmma_f32_16x16x32_f16 per tile give ~fp32 accuracy at f16 WMMA rate.
// Bandwidth-bound (~2.1 GB min HBM traffic -> ~90us @ 23.3 TB/s); masks are
// streamed twice (sums pass, GEMM pass w/ fused mask_row write). GEMM-pass
// mask loads and mask_row stores are non-temporal so the dead 164MB/800MB
// streams don't evict the L2-resident B operands from the 192MB L2.
// ---------------------------------------------------------------------------

typedef __attribute__((ext_vector_type(16))) _Float16 v16h;
typedef __attribute__((ext_vector_type(8)))  _Float16 v8h;
typedef __attribute__((ext_vector_type(8)))  float    v8f;
typedef __attribute__((ext_vector_type(4)))  float    v4f;

#define D_DIM   128
#define KCHUNK  64
#define LDA_SH  72   // f16 stride: 144B -> bank-conflict-free across 16 lanes

// ---------------- reductions ----------------

__global__ void row_sum_kernel(const float* __restrict__ A,
                               float* __restrict__ out, int cols) {
  __shared__ float red[256];
  const int row = blockIdx.x;
  const float* p = A + (size_t)row * cols;
  float s = 0.0f;
  for (int c = threadIdx.x; c < cols; c += 256) s += p[c];
  red[threadIdx.x] = s;
  __syncthreads();
  for (int off = 128; off > 0; off >>= 1) {
    if ((int)threadIdx.x < off) red[threadIdx.x] += red[threadIdx.x + off];
    __syncthreads();
  }
  if (threadIdx.x == 0) out[row] = red[0];
}

__global__ void col_sum_kernel(const float* __restrict__ A,
                               float* __restrict__ out, int rows, int cols) {
  const int col = blockIdx.x * blockDim.x + threadIdx.x;
  if (col >= cols) return;
  float s = 0.0f;
  for (int r = 0; r < rows; ++r) s += A[(size_t)r * cols + col];
  out[col] = s;
}

// ---------------- B prep: col-scale, f16 hi/lo split, transpose to [N][K] ---

__global__ void prep_bt_kernel(const float* __restrict__ E,
                               const float* __restrict__ colsum, int K,
                               _Float16* __restrict__ hi,
                               _Float16* __restrict__ lo) {
  const int idx = blockIdx.x * blockDim.x + threadIdx.x;
  if (idx >= K * D_DIM) return;
  const int k = idx / D_DIM;
  const int n = idx - k * D_DIM;
  const float v = E[idx] * rsqrtf(colsum[k]);
  const _Float16 h = (_Float16)v;
  const _Float16 l = (_Float16)(v - (float)h);
  hi[(size_t)n * K + k] = h;
  lo[(size_t)n * K + k] = l;
}

// ---------------- WMMA GEMM -------------------------------------------------
// Block = 128 threads (4 waves). Block owns rows [r0, r0+16) and all D=128
// columns; wave w owns N-tiles {32w, 32w+16}. A tile (16 x KCHUNK) staged in
// LDS as f16 once per chunk -> mask stripe read once per block. mask_row write
// fused into the A loader (templated). K loop split into an unguarded main
// body and one guarded tail chunk (K % 64 != 0 but K % 16 == 0).
//
// Fragment layouts per CDNA5 ISA 7.12.2 (wave32):
//   A f16 16x32: lane(M=lane&15, half=lane>>4); elems 0..7  = K kc+8*half+i,
//                elems 8..15 = K kc+16+8*half+i  (contiguous 8-f16 segments)
//   B f16 32x16: lane(N=lane&15, half); same K order as A (8-f16 segments)
//   C/D f32 16x16: elem v -> M = v + 8*half, N = lane&15

__device__ __forceinline__ v16h combine16(v8h a, v8h b) {
  v16h r;
#pragma unroll
  for (int i = 0; i < 8; ++i) { r[i] = a[i]; r[8 + i] = b[i]; }
  return r;
}

template <bool MASKROW, bool GUARD>
__device__ __forceinline__ void do_chunk(
    const float* __restrict__ A, int K, int k0, int r0, int lrow, int lk,
    float invRow, const _Float16* __restrict__ BThi,
    const _Float16* __restrict__ BTlo, _Float16* smA, int wave, int half,
    int ln, float* __restrict__ maskRowOut, v8f& acc0, v8f& acc1) {
  // ---- stage A stripe (fp32 -> f16 LDS), optional fused mask_row write ----
  const int kseg = k0 + lk;
  v4f f0 = {};
  v4f f1 = {};
  if (!GUARD || kseg < K) {  // K % 8 == 0 -> whole 8-float segment in/out
    const float* ap = A + (size_t)(r0 + lrow) * K + kseg;
    f0 = __builtin_nontemporal_load((const v4f*)ap);        // streamed once
    f1 = __builtin_nontemporal_load((const v4f*)(ap + 4));
    if (MASKROW) {
      float* mp = maskRowOut + (size_t)(r0 + lrow) * K + kseg;
      __builtin_nontemporal_store(f0 * invRow, (v4f*)mp);   // streamed out
      __builtin_nontemporal_store(f1 * invRow, (v4f*)(mp + 4));
    }
  }
  __syncthreads();  // protect previous chunk's LDS reads
  {
    _Float16* sp = smA + lrow * LDA_SH + lk;
#pragma unroll
    for (int i = 0; i < 4; ++i) sp[i] = (_Float16)f0[i];
#pragma unroll
    for (int i = 0; i < 4; ++i) sp[4 + i] = (_Float16)f1[i];
  }
  __syncthreads();

  // ---- compute: two 32-deep sub-chunks ----
#pragma unroll
  for (int kc = 0; kc < KCHUNK; kc += 32) {
    const _Float16* ar = smA + ln * LDA_SH + kc + half * 8;
    const v8h a0 = *(const v8h*)ar;
    const v8h a1 = *(const v8h*)(ar + 16);
    const v16h a = combine16(a0, a1);

#pragma unroll
    for (int t = 0; t < 2; ++t) {
      const int nn   = wave * 32 + t * 16 + ln;
      const int kb   = k0 + kc + half * 8;
      const size_t o = (size_t)nn * K + kb;
      v8h bh0 = {}, bh1 = {}, bl0 = {}, bl1 = {};
      if (!GUARD || kb < K)     { bh0 = *(const v8h*)(BThi + o);
                                  bl0 = *(const v8h*)(BTlo + o); }
      if (!GUARD || kb + 16 < K) { bh1 = *(const v8h*)(BThi + o + 16);
                                  bl1 = *(const v8h*)(BTlo + o + 16); }
      const v16h bh = combine16(bh0, bh1);
      const v16h bl = combine16(bl0, bl1);
      if (t == 0) {
        acc0 = __builtin_amdgcn_wmma_f32_16x16x32_f16(
            false, a, false, bh, (short)0, acc0, false, false);
        acc0 = __builtin_amdgcn_wmma_f32_16x16x32_f16(
            false, a, false, bl, (short)0, acc0, false, false);
      } else {
        acc1 = __builtin_amdgcn_wmma_f32_16x16x32_f16(
            false, a, false, bh, (short)0, acc1, false, false);
        acc1 = __builtin_amdgcn_wmma_f32_16x16x32_f16(
            false, a, false, bl, (short)0, acc1, false, false);
      }
    }
  }
}

template <bool MASKROW>
__global__ __launch_bounds__(128)
void gemm_symnorm_wmma_kernel(const float* __restrict__ A, int K,
                              const _Float16* __restrict__ BThi,
                              const _Float16* __restrict__ BTlo,
                              const float* __restrict__ rowsum,
                              float* __restrict__ out,
                              float* __restrict__ maskRowOut) {
  __shared__ __align__(16) _Float16 smA[16 * LDA_SH];

  const int tid  = threadIdx.x;
  const int lane = tid & 31;
  const int wave = tid >> 5;
  const int half = lane >> 4;
  const int ln   = lane & 15;
  const int r0   = blockIdx.x * 16;

  // A-loader assignment: 8 threads per row, 8 consecutive floats per thread
  const int lrow = tid >> 3;
  const int lk   = (tid & 7) * 8;
  const float invRow = MASKROW ? (1.0f / rowsum[r0 + lrow]) : 0.0f;

  v8f acc0 = {};
  v8f acc1 = {};

  const int KMAIN = K - (K % KCHUNK);
  int k0 = 0;
  for (; k0 < KMAIN; k0 += KCHUNK)  // hot loop: no guards, no exec masking
    do_chunk<MASKROW, false>(A, K, k0, r0, lrow, lk, invRow, BThi, BTlo, smA,
                             wave, half, ln, maskRowOut, acc0, acc1);
  if (KMAIN < K)                    // single ragged tail chunk
    do_chunk<MASKROW, true>(A, K, KMAIN, r0, lrow, lk, invRow, BThi, BTlo,
                            smA, wave, half, ln, maskRowOut, acc0, acc1);

  // ---- epilogue: apply diag(rsqrt(rowsum)) and store ----
#pragma unroll
  for (int v = 0; v < 8; ++v) {
    const int m = r0 + v + half * 8;
    const float s = rsqrtf(rowsum[m]);
    out[(size_t)m * D_DIM + wave * 32 + ln]      = acc0[v] * s;
    out[(size_t)m * D_DIM + wave * 32 + 16 + ln] = acc1[v] * s;
  }
}

// ---------------- host-side orchestration ----------------------------------

static inline size_t align_up(size_t x, size_t a) { return (x + a - 1) & ~(a - 1); }

extern "C" void kernel_launch(void* const* d_in, const int* in_sizes, int n_in,
                              void* d_out, int out_size, void* d_ws, size_t ws_size,
                              hipStream_t stream) {
  const float* mask   = (const float*)d_in[0];  // [B, U]
  const float* embed  = (const float*)d_in[1];  // [U, 128]
  const float* maskN  = (const float*)d_in[2];  // [U, U2]
  const float* embedX = (const float*)d_in[3];  // [U2, 128]

  const int U  = in_sizes[1] / D_DIM;
  const int B  = in_sizes[0] / U;
  const int U2 = in_sizes[3] / D_DIM;

  // Workspace carve-up
  char* ws = (char*)d_ws;
  size_t off = 0;
  float* rowsum_mask  = (float*)(ws + off); off = align_up(off + (size_t)B  * 4, 256);
  float* colsum_mask  = (float*)(ws + off); off = align_up(off + (size_t)U  * 4, 256);
  float* rowsum_neigh = (float*)(ws + off); off = align_up(off + (size_t)U  * 4, 256);
  float* colsum_neigh = (float*)(ws + off); off = align_up(off + (size_t)U2 * 4, 256);
  _Float16* bt1_hi = (_Float16*)(ws + off); off = align_up(off + (size_t)U  * D_DIM * 2, 256);
  _Float16* bt1_lo = (_Float16*)(ws + off); off = align_up(off + (size_t)U  * D_DIM * 2, 256);
  _Float16* bt2_hi = (_Float16*)(ws + off); off = align_up(off + (size_t)U2 * D_DIM * 2, 256);
  _Float16* bt2_lo = (_Float16*)(ws + off); off = align_up(off + (size_t)U2 * D_DIM * 2, 256);
  (void)ws_size; (void)n_in; (void)out_size;

  // Output layout: to_feats [B,128] | to_feats_neigh [U,128] | mask_row [B,U]
  float* to_feats   = (float*)d_out;
  float* to_feats_n = to_feats + (size_t)B * D_DIM;
  float* mask_row   = to_feats_n + (size_t)U * D_DIM;

  // 1) degree sums (one streaming read of each mask; regular temporal so the
  //    164MB mask has a chance to still be L2-resident for the GEMM pass)
  row_sum_kernel<<<B, 256, 0, stream>>>(mask, rowsum_mask, U);
  col_sum_kernel<<<(U + 255) / 256, 256, 0, stream>>>(mask, colsum_mask, B, U);
  row_sum_kernel<<<U, 256, 0, stream>>>(maskN, rowsum_neigh, U2);
  col_sum_kernel<<<(U2 + 255) / 256, 256, 0, stream>>>(maskN, colsum_neigh, U, U2);

  // 2) col-scaled, hi/lo-split, N-major B operands (tiny, L2 resident)
  prep_bt_kernel<<<((size_t)U * D_DIM + 255) / 256, 256, 0, stream>>>(
      embed, colsum_mask, U, bt1_hi, bt1_lo);
  prep_bt_kernel<<<((size_t)U2 * D_DIM + 255) / 256, 256, 0, stream>>>(
      embedX, colsum_neigh, U2, bt2_hi, bt2_lo);

  // 3) WMMA GEMMs (second streaming read of each mask; mask_row fused)
  gemm_symnorm_wmma_kernel<true><<<B / 16, 128, 0, stream>>>(
      mask, U, bt1_hi, bt1_lo, rowsum_mask, to_feats, mask_row);
  gemm_symnorm_wmma_kernel<false><<<U / 16, 128, 0, stream>>>(
      maskN, U2, bt2_hi, bt2_lo, rowsum_neigh, to_feats_n, nullptr);
}
